// SurvPLE_28346784153751
// MI455X (gfx1250) — compile-verified, matrix operand order checked by reading, not run on hardware
//
#include <hip/hip_runtime.h>
#include <hip/hip_bf16.h>

// ---------------------------------------------------------------------------
// SurvPLE (Cox partial likelihood) loss for MI455X / gfx1250.
//
//   theta      = min(-y_hat, 10)
//   risk_sum_i = sum_j exp(theta_j) * [T_j >= T_i]        (n x n matvec)
//   loss       = -mean( (theta - log(risk_sum)) * E )
//
// The O(n^2) risk-set matvec is executed on the WMMA pipe:
//   A(16x32 f16) = indicator tile [T[j0+k] >= T[i0+p]]  (0/1, exact in f16)
//   B(32x16 f16) = exp(theta)[j0+k] broadcast over all 16 columns,
//                  split into f16 hi + lo parts for near-f32 accuracy
//   C(16x16 f32) accumulator -> after the j loop, column q of D holds the
//                  full risk sums for the wave's 16 i-rows (replicated).
// Two v_wmma_f32_16x16x32_f16 per 16x32 tile (hi + lo).
// ---------------------------------------------------------------------------

typedef __attribute__((ext_vector_type(16))) _Float16 v16h;
typedef __attribute__((ext_vector_type(8)))  _Float16 v8h;
typedef __attribute__((ext_vector_type(8)))  float    v8f;
typedef __attribute__((ext_vector_type(4)))  float    v4f;

#define SURV_CLIP 10.0f

// ---------------------------------------------------------------------------
// Kernel 1: elementwise prep.  theta = min(-y, 10); et = exp(theta);
// split et into f16 hi/lo so WMMA accumulation is accurate to ~2^-22 rel.
// ---------------------------------------------------------------------------
__global__ void surv_prep_kernel(const float* __restrict__ y_hat,
                                 float* __restrict__ theta,
                                 _Float16* __restrict__ ehi,
                                 _Float16* __restrict__ elo,
                                 int n) {
    int i = blockIdx.x * blockDim.x + threadIdx.x;
    if (i >= n) return;
    float th = fminf(-y_hat[i], SURV_CLIP);
    float et = expf(th);
    _Float16 hi = (_Float16)et;
    _Float16 lo = (_Float16)(et - (float)hi);
    theta[i] = th;
    ehi[i]   = hi;
    elo[i]   = lo;
}

// ---------------------------------------------------------------------------
// Kernel 2: WMMA risk-set matvec + per-wave loss partials.
// One wave per 16 consecutive i's; the wave sweeps all j in chunks of 32.
// blockDim.x must be 256 (8 waves); grid = n/128 blocks.
// ---------------------------------------------------------------------------
__global__ void surv_wmma_kernel(const float* __restrict__ T,
                                 const float* __restrict__ theta,
                                 const _Float16* __restrict__ ehi,
                                 const _Float16* __restrict__ elo,
                                 const int* __restrict__ E,
                                 float* __restrict__ partials,
                                 int n) {
    const int lane  = threadIdx.x & 31;
    const int wave  = threadIdx.x >> 5;
    const int gwave = blockIdx.x * (blockDim.x >> 5) + wave;
    const int i0    = gwave << 4;          // 16 i-rows per wave
    const int h     = lane >> 4;           // lane half (selects K groups)
    const int p     = lane & 15;           // A-matrix row M = p

    const float Ti = T[i0 + p];

    const v4f* Tv   = (const v4f*)T;
    const v8h* EHv  = (const v8h*)ehi;
    const v8h* ELv  = (const v8h*)elo;

    v8f c = {};   // f32 accumulator: risk sums for i0..i0+15 (cols replicated)

    for (int j0 = 0; j0 < n; j0 += 32) {
        // Lane's A/B elements 0..7  cover K = j0 + 8h .. +7
        // Lane's A/B elements 8..15 cover K = j0 + 16 + 8h .. +7
        const int base1 = j0 + (h << 3);
        const int base2 = base1 + 16;

        // T values for this lane's 16 K slots (16B-aligned vector loads).
        v4f t0 = Tv[(base1 >> 2) + 0];
        v4f t1 = Tv[(base1 >> 2) + 1];
        v4f t2 = Tv[(base2 >> 2) + 0];
        v4f t3 = Tv[(base2 >> 2) + 1];

        // exp(theta) hi/lo, broadcast to all N columns (B is column-constant).
        v8h bh1 = EHv[base1 >> 3];
        v8h bh2 = EHv[base2 >> 3];
        v8h bl1 = ELv[base1 >> 3];
        v8h bl2 = ELv[base2 >> 3];

        v16h a, bh, bl;
        #pragma unroll
        for (int k = 0; k < 4; ++k) {
            a[k]      = (t0[k] >= Ti) ? (_Float16)1.0f : (_Float16)0.0f;
            a[4 + k]  = (t1[k] >= Ti) ? (_Float16)1.0f : (_Float16)0.0f;
            a[8 + k]  = (t2[k] >= Ti) ? (_Float16)1.0f : (_Float16)0.0f;
            a[12 + k] = (t3[k] >= Ti) ? (_Float16)1.0f : (_Float16)0.0f;
        }
        #pragma unroll
        for (int k = 0; k < 8; ++k) {
            bh[k]     = bh1[k];
            bh[8 + k] = bh2[k];
            bl[k]     = bl1[k];
            bl[8 + k] = bl2[k];
        }

        // D += A x B  (hi part, then lo part, same f32 accumulator)
        c = __builtin_amdgcn_wmma_f32_16x16x32_f16(
                false, a, false, bh, (short)0, c, false, false);
        c = __builtin_amdgcn_wmma_f32_16x16x32_f16(
                false, a, false, bl, (short)0, c, false, false);
    }

    // D layout (16x16 f32): lane with N = lane&15, VGPR r holds row M = r + 8h.
    // All columns are identical; use column 0 of each half (lanes 0 and 16).
    if (p == 0) {
        float s = 0.0f;
        #pragma unroll
        for (int r = 0; r < 8; ++r) {
            int i = i0 + (h << 3) + r;
            s += (theta[i] - logf(c[r])) * (float)E[i];
        }
        partials[(gwave << 1) + h] = s;
    }
}

// ---------------------------------------------------------------------------
// Kernel 3: deterministic fixed-order reduction of per-wave partials.
// ---------------------------------------------------------------------------
__global__ void surv_reduce_kernel(const float* __restrict__ partials,
                                   int m, int n, float* __restrict__ out) {
    __shared__ float sdata[256];
    float s = 0.0f;
    for (int t = threadIdx.x; t < m; t += 256) s += partials[t];
    sdata[threadIdx.x] = s;
    __syncthreads();
    for (int off = 128; off > 0; off >>= 1) {
        if ((int)threadIdx.x < off) sdata[threadIdx.x] += sdata[threadIdx.x + off];
        __syncthreads();
    }
    if (threadIdx.x == 0) out[0] = -sdata[0] / (float)n;
}

// ---------------------------------------------------------------------------
extern "C" void kernel_launch(void* const* d_in, const int* in_sizes, int n_in,
                              void* d_out, int out_size, void* d_ws, size_t ws_size,
                              hipStream_t stream) {
    const float* y_hat = (const float*)d_in[0];
    const float* T     = (const float*)d_in[1];
    const int*   E     = (const int*)d_in[2];
    float*       out   = (float*)d_out;
    const int n = in_sizes[0];      // 16384; assumed multiple of 128

    // Workspace layout: theta[n] f32 | ehi[n] f16 | elo[n] f16 | partials f32
    float*    theta    = (float*)d_ws;
    _Float16* ehi      = (_Float16*)(theta + n);
    _Float16* elo      = ehi + n;
    float*    partials = (float*)(elo + n);
    const int nwaves   = n >> 4;          // 16 i's per wave
    const int nparts   = nwaves << 1;     // 2 partials per wave (lane 0 & 16)

    surv_prep_kernel<<<(n + 255) / 256, 256, 0, stream>>>(y_hat, theta, ehi, elo, n);

    // 8 waves / block, 16 i's / wave -> 128 i's per block.
    surv_wmma_kernel<<<n / 128, 256, 0, stream>>>(T, theta, ehi, elo, E, partials, n);

    surv_reduce_kernel<<<1, 256, 0, stream>>>(partials, nparts, n, out);
}
